// OcclusionAttention_65137474011194
// MI455X (gfx1250) — compile-verified, matrix operand order checked by reading, not run on hardware
//
#include <hip/hip_runtime.h>

// ---------------------------------------------------------------------------
// OcclusionAttention on MI455X (gfx1250, wave32, WMMA)
//   B=16, N=2048, D=1024, P=APP=1024, M = B*N = 32768
//   out[b,n,a] = att[b,n] * relu(x·Wv^T + bv)[b,n,a]
//   att[b,n]   = rnorm[b,n] * <xp[b,n,:], ws[b,:]> / msum[b]
//   ws[b,p]    = sum_n mask[n,b] * rnorm[b,n] * xp[b,n,p]
//   xp         = relu(x·Wp^T + bp)   (stored bf16), rnorm = 1/(||xp||+eps)
// GEMMs: v_wmma_f32_16x16x32_bf16 with async global->LDS double buffering
// (ASYNCcnt), hand-unrolled x2 so buffer indices are compile-time constants.
// ---------------------------------------------------------------------------

typedef __attribute__((ext_vector_type(16))) __bf16 v16bf;
typedef __attribute__((ext_vector_type(8)))  float  v8f;
typedef __attribute__((ext_vector_type(4)))  int    v4i_;

#define B_   16
#define N_   2048
#define D_   1024
#define P_   1024
#define MT   (B_ * N_)       // 32768 rows
#define EPSV 1e-8f

#if __has_builtin(__builtin_amdgcn_global_load_async_to_lds_b128) && \
    __has_builtin(__builtin_amdgcn_s_wait_asynccnt)
#define USE_ASYNC_LDS 1
#else
#define USE_ASYNC_LDS 0
#endif

// ---------------- fp32 -> bf16 conversion ----------------------------------
__global__ void k_cvt_bf16(const float* __restrict__ src,
                           __bf16* __restrict__ dst, int n) {
  int i = (blockIdx.x * blockDim.x + threadIdx.x) * 4;
  if (i + 3 < n) {
    float4 f = *(const float4*)(src + i);
    dst[i + 0] = (__bf16)f.x;
    dst[i + 1] = (__bf16)f.y;
    dst[i + 2] = (__bf16)f.z;
    dst[i + 3] = (__bf16)f.w;
  } else {
    for (int j = i; j < n; ++j) dst[j] = (__bf16)src[j];
  }
}

// ---------------- msum[b] = sum_n mask[n,b] --------------------------------
__global__ void k_msum(const float* __restrict__ mask, float* __restrict__ msum) {
  __shared__ float red[256];
  int b = blockIdx.x;
  float s = 0.f;
  for (int n = threadIdx.x; n < N_; n += 256) s += mask[n * B_ + b];
  red[threadIdx.x] = s;
  __syncthreads();
  for (int off = 128; off; off >>= 1) {
    if ((int)threadIdx.x < off) red[threadIdx.x] += red[threadIdx.x + off];
    __syncthreads();
  }
  if (threadIdx.x == 0) msum[b] = red[0];
}

// ---------------- bf16 WMMA GEMM:  C = relu(A * W^T + bias) ----------------
// A: [MT][D_] bf16 row-major, W: [P_][D_] bf16 row-major.
// MODE 0: store bf16 xp.  MODE 1: nontemporal f32 out * scale[row].
#define BM 128
#define BN 128
#define BK 32
#define KT_ (D_ / BK)   // 32 K-tiles (even)
#define LDT (BK + 16)   // 48 elems: rows stay 16B-aligned, banks spread

#if USE_ASYNC_LDS
__device__ __forceinline__ void async_cp16(const __bf16* g, __bf16* l) {
  // arg0: global (AS1) int4*, arg1: LDS (AS3) int4*; generic-LDS pointer's
  // low 32 bits are the LDS byte offset (aperture rules)
  __builtin_amdgcn_global_load_async_to_lds_b128(
      (__attribute__((address_space(1))) v4i_*)(uintptr_t)g,
      (__attribute__((address_space(3))) v4i_*)(unsigned)(uintptr_t)l,
      0, 0);
}
#endif

template <int MODE>
__global__ __launch_bounds__(256) void k_gemm(
    const __bf16* __restrict__ A, const __bf16* __restrict__ W,
    const float* __restrict__ bias, const float* __restrict__ scale,
    __bf16* __restrict__ outb, float* __restrict__ outf) {
  const int m0   = blockIdx.x * BM;
  const int n0   = blockIdx.y * BN;
  const int t    = threadIdx.x;
  const int wave = t >> 5;
  const int lane = t & 31;
  const int half = lane >> 4;   // K-group select (ISA 16-bit A/B layout)
  const int r16  = lane & 15;   // M (A) / N (B) row within fragment
  const int wy   = wave >> 2;   // 2 waves over M  (64 rows each)
  const int wx   = wave & 3;    // 4 waves over N  (32 cols each)

  v8f zero = {};
  v8f acc[4][2];
#pragma unroll
  for (int i = 0; i < 4; ++i)
#pragma unroll
    for (int j = 0; j < 2; ++j) acc[i][j] = zero;

  union Frag { uint4 u[2]; v16bf v; };

#if USE_ASYNC_LDS
  // ---- double-buffered LDS, async global->LDS (no VGPR staging) ----
  __shared__ __bf16 As[2][BM][LDT];
  __shared__ __bf16 Bs[2][BM][LDT];

  // issue one tile pair into buffer `buf` (4 async b128 per thread)
  auto issue_tile = [&](int buf, int kk) {
#pragma unroll
    for (int s = 0; s < 2; ++s) {
      int idx = t + s * 256;
      int row = idx >> 2;
      int q   = idx & 3;
      async_cp16(A + (size_t)(m0 + row) * D_ + kk + q * 8, &As[buf][row][q * 8]);
      async_cp16(W + (size_t)(n0 + row) * D_ + kk + q * 8, &Bs[buf][row][q * 8]);
    }
  };

  // consume buffer `buf` (constant at each call site)
  auto compute_tile = [&](const __bf16 (*Asb)[LDT], const __bf16 (*Bsb)[LDT]) {
    Frag a[4], bf[2];
#pragma unroll
    for (int mf = 0; mf < 4; ++mf) {
      const __bf16* p = &Asb[wy * 64 + mf * 16 + r16][0];
      a[mf].u[0] = *(const uint4*)(p + half * 8);        // K 0-7 / 8-15
      a[mf].u[1] = *(const uint4*)(p + 16 + half * 8);   // K 16-23 / 24-31
    }
#pragma unroll
    for (int nf = 0; nf < 2; ++nf) {
      const __bf16* p = &Bsb[wx * 32 + nf * 16 + r16][0];
      bf[nf].u[0] = *(const uint4*)(p + half * 8);
      bf[nf].u[1] = *(const uint4*)(p + 16 + half * 8);
    }
#pragma unroll
    for (int mf = 0; mf < 4; ++mf)
#pragma unroll
      for (int nf = 0; nf < 2; ++nf)
        acc[mf][nf] = __builtin_amdgcn_wmma_f32_16x16x32_bf16(
            false, a[mf].v, false, bf[nf].v, (short)0, acc[mf][nf],
            false, false);
  };

  issue_tile(0, 0);
#pragma unroll 1
  for (int kt = 0; kt < KT_; kt += 2) {
    // ---- buffer 0 ----
    if (kt + 1 < KT_) {
      issue_tile(1, (kt + 1) * BK);
      __builtin_amdgcn_s_wait_asynccnt(4);  // tile kt complete
    } else {
      __builtin_amdgcn_s_wait_asynccnt(0);
    }
    __syncthreads();
    compute_tile(As[0], Bs[0]);
    __syncthreads();                         // buffer-0 reuse guard
    // ---- buffer 1 ----
    if (kt + 2 < KT_) {
      issue_tile(0, (kt + 2) * BK);
      __builtin_amdgcn_s_wait_asynccnt(4);  // tile kt+1 complete
    } else {
      __builtin_amdgcn_s_wait_asynccnt(0);
    }
    __syncthreads();
    compute_tile(As[1], Bs[1]);
    __syncthreads();                         // buffer-1 reuse guard
  }
#else
  // ---- fallback: single-buffer LDS via VGPR staging ----
  __shared__ __bf16 As[BM][LDT];
  __shared__ __bf16 Bs[BM][LDT];

  for (int kk = 0; kk < D_; kk += BK) {
#pragma unroll
    for (int s = 0; s < 2; ++s) {
      int idx = t + s * 256;
      int row = idx >> 2;
      int q   = idx & 3;
      *(uint4*)&As[row][q * 8] =
          *(const uint4*)(A + (size_t)(m0 + row) * D_ + kk + q * 8);
      *(uint4*)&Bs[row][q * 8] =
          *(const uint4*)(W + (size_t)(n0 + row) * D_ + kk + q * 8);
    }
    __syncthreads();

    Frag a[4], bf[2];
#pragma unroll
    for (int mf = 0; mf < 4; ++mf) {
      const __bf16* p = &As[wy * 64 + mf * 16 + r16][0];
      a[mf].u[0] = *(const uint4*)(p + half * 8);
      a[mf].u[1] = *(const uint4*)(p + 16 + half * 8);
    }
#pragma unroll
    for (int nf = 0; nf < 2; ++nf) {
      const __bf16* p = &Bs[wx * 32 + nf * 16 + r16][0];
      bf[nf].u[0] = *(const uint4*)(p + half * 8);
      bf[nf].u[1] = *(const uint4*)(p + 16 + half * 8);
    }
#pragma unroll
    for (int mf = 0; mf < 4; ++mf)
#pragma unroll
      for (int nf = 0; nf < 2; ++nf)
        acc[mf][nf] = __builtin_amdgcn_wmma_f32_16x16x32_bf16(
            false, a[mf].v, false, bf[nf].v, (short)0, acc[mf][nf],
            false, false);
    __syncthreads();
  }
#endif

  // epilogue: C VGPR i -> M = i + 8*half (lanes split), N = r16
#pragma unroll
  for (int mf = 0; mf < 4; ++mf) {
#pragma unroll
    for (int nf = 0; nf < 2; ++nf) {
      const int gcol = n0 + wx * 32 + nf * 16 + r16;
      const float bb = bias[gcol];
#pragma unroll
      for (int i = 0; i < 8; ++i) {
        const int grow = m0 + wy * 64 + mf * 16 + half * 8 + i;
        float v = acc[mf][nf][i] + bb;
        v = v > 0.f ? v : 0.f;
        if (MODE == 0) {
          outb[(size_t)grow * P_ + gcol] = (__bf16)v;   // stays hot in L2
        } else {
          // streamed once, never re-read: bypass temporal caching
          __builtin_nontemporal_store(v * scale[grow],
                                      &outf[(size_t)grow * P_ + gcol]);
        }
      }
    }
  }
}

// ---------------- per-row 1/(||xp||+eps), wave per row ---------------------
__global__ void k_rownorm(const __bf16* __restrict__ xp,
                          float* __restrict__ rnorm) {
  const int row  = blockIdx.x * 8 + (threadIdx.x >> 5);
  const int lane = threadIdx.x & 31;
  const __bf16* p = xp + (size_t)row * P_;
  float s = 0.f;
#pragma unroll
  for (int j = 0; j < P_; j += 256) {
    union { uint4 u; __bf16 h[8]; } v;
    v.u = *(const uint4*)(p + j + lane * 8);
#pragma unroll
    for (int e = 0; e < 8; ++e) { float f = (float)v.h[e]; s += f * f; }
  }
  for (int off = 16; off; off >>= 1) s += __shfl_xor(s, off, 32);
  if (lane == 0) rnorm[row] = 1.f / (sqrtf(s) + EPSV);
}

// ---------------- ws[b,p] = sum_n mask[n,b]*rnorm[b,n]*xp[b,n,p] -----------
__global__ void k_ws(const __bf16* __restrict__ xp,
                     const float* __restrict__ mask,
                     const float* __restrict__ rnorm,
                     float* __restrict__ wsv) {
  const int b    = blockIdx.y;
  const int pidx = blockIdx.x * 256 + threadIdx.x;
  const __bf16* base = xp + (size_t)b * N_ * P_ + pidx;
  float acc = 0.f;
  for (int n = 0; n < N_; ++n) {
    __builtin_prefetch(base + (size_t)(n + 8) * P_, 0, 0);
    const float w = mask[n * B_ + b] * rnorm[b * N_ + n];
    acc += w * (float)base[(size_t)n * P_];
  }
  wsv[b * P_ + pidx] = acc;
}

// ---------------- att[row] = rnorm*<xp_row, ws_b>/msum[b], wave/row --------
__global__ void k_att(const __bf16* __restrict__ xp,
                      const float* __restrict__ wsv,
                      const float* __restrict__ rnorm,
                      const float* __restrict__ msum,
                      float* __restrict__ att) {
  const int row  = blockIdx.x * 8 + (threadIdx.x >> 5);
  const int lane = threadIdx.x & 31;
  const int b    = row >> 11;   // row / N_
  const __bf16* p = xp + (size_t)row * P_;
  const float*  w = wsv + b * P_;
  float s = 0.f;
  for (int j = lane; j < P_; j += 32) s += (float)p[j] * w[j];
  for (int off = 16; off; off >>= 1) s += __shfl_xor(s, off, 32);
  if (lane == 0) att[row] = rnorm[row] * s / msum[b];
}

// ---------------------------------------------------------------------------
extern "C" void kernel_launch(void* const* d_in, const int* in_sizes, int n_in,
                              void* d_out, int out_size, void* d_ws,
                              size_t ws_size, hipStream_t stream) {
  const float* x    = (const float*)d_in[0];  // [B,N,D]
  const float* mask = (const float*)d_in[1];  // [N,B]
  const float* Wp   = (const float*)d_in[2];  // [P,D]
  const float* bp   = (const float*)d_in[3];  // [P]
  const float* Wv   = (const float*)d_in[4];  // [P,D]
  const float* bv   = (const float*)d_in[5];  // [P]
  float* out = (float*)d_out;                 // [B,N,P] f32

  char* w = (char*)d_ws;
  __bf16* xb    = (__bf16*)w; w += (size_t)MT * D_ * 2;   // 64 MiB
  __bf16* wpb   = (__bf16*)w; w += (size_t)P_ * D_ * 2;   // 2 MiB
  __bf16* wvb   = (__bf16*)w; w += (size_t)P_ * D_ * 2;   // 2 MiB
  __bf16* xp    = (__bf16*)w; w += (size_t)MT * P_ * 2;   // 64 MiB
  float*  rnorm = (float*)w;  w += (size_t)MT * 4;
  float*  msum  = (float*)w;  w += 256;
  float*  wsv   = (float*)w;  w += (size_t)B_ * P_ * 4;
  float*  att   = (float*)w;  w += (size_t)MT * 4;

  k_cvt_bf16<<<(MT * D_ / 4 + 255) / 256, 256, 0, stream>>>(x, xb, MT * D_);
  k_cvt_bf16<<<(P_ * D_ / 4 + 255) / 256, 256, 0, stream>>>(Wp, wpb, P_ * D_);
  k_cvt_bf16<<<(P_ * D_ / 4 + 255) / 256, 256, 0, stream>>>(Wv, wvb, P_ * D_);
  k_msum<<<B_, 256, 0, stream>>>(mask, msum);

  dim3 gg(MT / BM, P_ / BN);  // 256 x 8
  k_gemm<0><<<gg, 256, 0, stream>>>(xb, wpb, bp, nullptr, xp, nullptr);

  k_rownorm<<<MT / 8, 256, 0, stream>>>(xp, rnorm);
  k_ws<<<dim3(P_ / 256, B_), 256, 0, stream>>>(xp, mask, rnorm, wsv);
  k_att<<<MT / 8, 256, 0, stream>>>(xp, wsv, rnorm, msum, att);

  k_gemm<1><<<gg, 256, 0, stream>>>(xb, wvb, bv, att, nullptr, out);
}
